// NetworkAdditive_34359738846
// MI455X (gfx1250) — compile-verified, matrix operand order checked by reading, not run on hardware
//
#include <hip/hip_runtime.h>
#include <math.h>

// Problem constants (from the reference)
#define T_ 512
#define B_ 64
#define N_ 256
#define H_ 64
#define D_ 128
#define P_ 64
#define M_ (B_*N_)      // 16384 rows of the recurrent GEMM (b,n flattened)
#define TB_ (T_*B_)     // 32768 rows of the front/back GEMMs (t,b flattened)

typedef __bf16 bf16_t;
typedef __attribute__((ext_vector_type(16))) __bf16 v16bf;
typedef __attribute__((ext_vector_type(8)))  float  v8f;

__device__ __forceinline__ v8f wmma_bf16(v16bf a, v16bf b, v8f c) {
  // D = A(16x32 bf16) x B(32x16 bf16) + C(16x16 f32)
  return __builtin_amdgcn_wmma_f32_16x16x32_bf16(false, a, false, b, (short)0, c,
                                                 false, false);
}

// 16x32 bf16 A/B operand per CDNA5 layout:
// lane r = row(A)/col(B) in [0,16), hi = lane>>4;
// VGPRs 0-3 hold K = kb + hi*8 + {0..7}; VGPRs 4-7 hold K = kb + 16 + hi*8 + {0..7}
__device__ __forceinline__ v16bf frag_bf16(const bf16_t* base, int ld, int lane, int kb) {
  const int r = lane & 15, hi = (lane >> 4) & 1;
  const bf16_t* p = base + (size_t)r * ld;
  v16bf v;
#pragma unroll
  for (int i = 0; i < 8; ++i) v[i] = p[kb + hi * 8 + i];
#pragma unroll
  for (int i = 0; i < 8; ++i) v[8 + i] = p[kb + 16 + hi * 8 + i];
  return v;
}

__device__ __forceinline__ v16bf frag_f32(const float* base, int ld, int lane, int kb) {
  const int r = lane & 15, hi = (lane >> 4) & 1;
  const float* p = base + (size_t)r * ld;
  v16bf v;
#pragma unroll
  for (int i = 0; i < 8; ++i) v[i] = (bf16_t)p[kb + hi * 8 + i];
#pragma unroll
  for (int i = 0; i < 8; ++i) v[8 + i] = (bf16_t)p[kb + 16 + hi * 8 + i];
  return v;
}

__device__ __forceinline__ float sigm(float x) {
  return 1.f / (1.f + __expf(-x));        // v_exp_f32 + rcp
}
__device__ __forceinline__ float tanh_f(float x) {
  const float e = __expf(-2.f * x);
  return (1.f - e) / (1.f + e);
}

// ---------------------------------------------------------------------------
// Kernel 1: x2n[t,b,n] = input[t,b,:] . W2n[n,:] + b2n[n]
// M = TB_ = 32768, K = D_ = 128, N = N_ = 256. 8 waves/block, 16 rows/wave.
// ---------------------------------------------------------------------------
__global__ __launch_bounds__(256) void x2n_gemm_kernel(
    const float* __restrict__ input, const float* __restrict__ W2n,
    const float* __restrict__ b2n, float* __restrict__ x2n) {
  const int lane = threadIdx.x & 31;
  const int wave = threadIdx.x >> 5;
  const int mrow = blockIdx.x * 128 + wave * 16;
  const int lcol = lane & 15, hi = lane >> 4;

  v16bf afr[4];
#pragma unroll
  for (int k = 0; k < 4; ++k)
    afr[k] = frag_f32(input + (size_t)mrow * D_, D_, lane, k * 32);

  for (int nt = 0; nt < 16; ++nt) {
    v8f acc;
    const float bv = b2n[nt * 16 + lcol];
#pragma unroll
    for (int j = 0; j < 8; ++j) acc[j] = bv;
#pragma unroll
    for (int k = 0; k < 4; ++k) {
      v16bf bfr = frag_f32(W2n + (size_t)(nt * 16) * D_, D_, lane, k * 32);
      acc = wmma_bf16(afr[k], bfr, acc);
    }
    // C layout: VGPR j -> M = hi*8 + j, N = lcol
    const int m = mrow + hi * 8;
    const int n = nt * 16 + lcol;
#pragma unroll
    for (int j = 0; j < 8; ++j) x2n[(size_t)(m + j) * N_ + n] = acc[j];
  }
}

// ---------------------------------------------------------------------------
// Kernel 2: the T=512 LSTM scan. Each wave owns a 16-row M-tile (b,n pairs);
// W_hh lives in LDS (bf16), h relayouts through LDS each step, c/gain state in
// registers. Per step per wave: 4 gate-groups x 4 h-subtiles x 2 K-halves = 32
// v_wmma_f32_16x16x32_bf16.
// ---------------------------------------------------------------------------
__global__ __launch_bounds__(256) void lstm_scan_kernel(
    const float* __restrict__ x2n, const float* __restrict__ W_ih,
    const float* __restrict__ W_hh, const float* __restrict__ b_ih,
    const float* __restrict__ b_hh, const float* __restrict__ Wg,
    const float* __restrict__ bg, bf16_t* __restrict__ outn) {
  __shared__ bf16_t whh[4 * H_ * H_];   // 256x64 bf16 = 32 KB, [gate][h] (== B rows)
  __shared__ bf16_t hbuf[128 * H_];     // 16 KB, [m_local][h]
  __shared__ float  xbuf[128];

  const int tid = threadIdx.x;
  const int lane = tid & 31;
  const int wave = tid >> 5;
  const int m0 = blockIdx.x * 128;
  const int mw = wave * 16;
  const int lcol = lane & 15, hi = lane >> 4;

  for (int i = tid; i < 4 * H_ * H_; i += 256) whh[i] = (bf16_t)W_hh[i];
  for (int i = tid; i < 128 * H_; i += 256) hbuf[i] = (bf16_t)0.f;

  // Per-lane column constants: nt = gt*4 + ht covers gate column nt*16 + lcol
  float biasc[16], wihc[16];
#pragma unroll
  for (int nt = 0; nt < 16; ++nt) {
    const int c = nt * 16 + lcol;
    biasc[nt] = b_ih[c] + b_hh[c];
    wihc[nt] = W_ih[c];                 // W_ih[:,0]
  }
  float wgc[4];
#pragma unroll
  for (int ht = 0; ht < 4; ++ht) wgc[ht] = Wg[ht * 16 + lcol];
  const float bg0 = bg[0];

  float cst[4][8];                      // c state, C-layout [h-subtile][row j]
#pragma unroll
  for (int a = 0; a < 4; ++a)
#pragma unroll
    for (int j = 0; j < 8; ++j) cst[a][j] = 0.f;
  float g1[8], g2[8];                   // g_{t-1}, g_{t-2} per owned row
#pragma unroll
  for (int j = 0; j < 8; ++j) { g1[j] = 0.f; g2[j] = 0.f; }

  const bf16_t* hrow = hbuf + mw * H_;
  __syncthreads();

  for (int t = 0; t < T_; ++t) {
    if (tid < 128) {
      xbuf[tid] = x2n[(size_t)t * M_ + m0 + tid];
      if (t + 1 < T_)                    // global_prefetch_b8 for next step's row
        __builtin_prefetch(&x2n[(size_t)(t + 1) * M_ + m0 + tid], 0, 1);
    }
    __syncthreads();

    float xr[8];
#pragma unroll
    for (int j = 0; j < 8; ++j) xr[j] = xbuf[mw + hi * 8 + j];

    v16bf alo = frag_bf16(hrow, H_, lane, 0);    // h tile, K = 0..31
    v16bf ahi = frag_bf16(hrow, H_, lane, 32);   // h tile, K = 32..63

    float gpart[8];
#pragma unroll
    for (int j = 0; j < 8; ++j) gpart[j] = 0.f;

#pragma unroll
    for (int ht = 0; ht < 4; ++ht) {
      v8f acc[4];
#pragma unroll
      for (int gt = 0; gt < 4; ++gt) {           // i, f, c~, o
        const int nt = gt * 4 + ht;
        v8f a;
#pragma unroll
        for (int j = 0; j < 8; ++j) a[j] = biasc[nt] + wihc[nt] * xr[j];
        v16bf blo = frag_bf16(whh + nt * 16 * H_, H_, lane, 0);
        v16bf bhi = frag_bf16(whh + nt * 16 * H_, H_, lane, 32);
        a = wmma_bf16(alo, blo, a);
        a = wmma_bf16(ahi, bhi, a);
        acc[gt] = a;
      }
#pragma unroll
      for (int j = 0; j < 8; ++j) {
        const float ig = sigm(acc[0][j]);
        const float fg = sigm(acc[1][j]);
        const float cg = tanh_f(acc[2][j]);
        const float og = sigm(acc[3][j]);
        const float c = fg * cst[ht][j] + ig * cg;
        cst[ht][j] = c;
        const float h = og * tanh_f(c);
        gpart[j] += h * wgc[ht];
        hbuf[(mw + hi * 8 + j) * H_ + ht * 16 + lcol] = (bf16_t)h;  // C->A relayout
      }
    }

    // g_t[m] = sum_h h[m,h]*wg[h] (+bg): reduce across the 16-lane column group
#pragma unroll
    for (int s = 1; s < 16; s <<= 1) {
#pragma unroll
      for (int j = 0; j < 8; ++j) gpart[j] += __shfl_xor(gpart[j], s, 32);
    }

    float gain[8];
#pragma unroll
    for (int j = 0; j < 8; ++j) {
      const float gt_ = gpart[j] + bg0;
      gain[j] = (t == 0) ? gt_ : ((t == 1) ? 1.f : g2[j]);
      g2[j] = g1[j];
      g1[j] = gt_;
    }
    if (lcol < 8) {                               // one writer per owned row
      const int j = lcol;
      const float z = xr[j] - 1.f;
      const float sp = (z > 20.f) ? z : log1pf(__expf(z));
      outn[(size_t)t * M_ + m0 + mw + hi * 8 + j] = (bf16_t)(gain[j] * sp);
    }
    __syncthreads();                              // protect xbuf WAR
  }
}

// ---------------------------------------------------------------------------
// Kernel 3: out[t,b,p] = out_neurons[t,b,:] . Wout[p,:] + bout[p]
// M = TB_ = 32768, K = N_ = 256, N = P_ = 64.
// ---------------------------------------------------------------------------
__global__ __launch_bounds__(256) void out_gemm_kernel(
    const bf16_t* __restrict__ outn, const float* __restrict__ Wout,
    const float* __restrict__ bout, float* __restrict__ out) {
  const int lane = threadIdx.x & 31;
  const int wave = threadIdx.x >> 5;
  const int mrow = blockIdx.x * 128 + wave * 16;
  const int lcol = lane & 15, hi = lane >> 4;

  v8f acc[4];
#pragma unroll
  for (int p = 0; p < 4; ++p) {
    const float bv = bout[p * 16 + lcol];
#pragma unroll
    for (int j = 0; j < 8; ++j) acc[p][j] = bv;
  }
  for (int k = 0; k < 8; ++k) {
    v16bf af = frag_bf16(outn + (size_t)mrow * N_, N_, lane, k * 32);
#pragma unroll
    for (int p = 0; p < 4; ++p) {
      v16bf bf2 = frag_f32(Wout + (size_t)(p * 16) * N_, N_, lane, k * 32);
      acc[p] = wmma_bf16(af, bf2, acc[p]);
    }
  }
  const int m = mrow + hi * 8;
#pragma unroll
  for (int p = 0; p < 4; ++p)
#pragma unroll
    for (int j = 0; j < 8; ++j)
      out[(size_t)(m + j) * P_ + p * 16 + lcol] = acc[p][j];
}

// ---------------------------------------------------------------------------
extern "C" void kernel_launch(void* const* d_in, const int* in_sizes, int n_in,
                              void* d_out, int out_size, void* d_ws, size_t ws_size,
                              hipStream_t stream) {
  const float* input = (const float*)d_in[0];
  const float* W2n  = (const float*)d_in[1];
  const float* b2n  = (const float*)d_in[2];
  const float* W_ih = (const float*)d_in[3];
  const float* W_hh = (const float*)d_in[4];
  const float* b_ih = (const float*)d_in[5];
  const float* b_hh = (const float*)d_in[6];
  const float* Wg   = (const float*)d_in[7];
  const float* bg   = (const float*)d_in[8];
  const float* Wout = (const float*)d_in[9];
  const float* bout = (const float*)d_in[10];

  // Workspace: x2n f32 (33.5 MB) | out_neurons bf16 (16.8 MB)
  float*  x2n  = (float*)d_ws;
  bf16_t* outn = (bf16_t*)((char*)d_ws + (size_t)TB_ * N_ * sizeof(float));

  x2n_gemm_kernel<<<TB_ / 128, 256, 0, stream>>>(input, W2n, b2n, x2n);
  lstm_scan_kernel<<<M_ / 128, 256, 0, stream>>>(x2n, W_ih, W_hh, b_ih, b_hh,
                                                 Wg, bg, outn);
  out_gemm_kernel<<<TB_ / 128, 256, 0, stream>>>(outn, Wout, bout, (float*)d_out);
}
